// UpFlowNetwork_45380624450252
// MI455X (gfx1250) — matrix operand order repaired and checked by reading, not compiled, not run on hardware
//
#include <hip/hip_runtime.h>

// ---------------- types ----------------
typedef __attribute__((ext_vector_type(8)))  unsigned short us8;
typedef __attribute__((ext_vector_type(16))) unsigned short us16;
typedef __attribute__((ext_vector_type(16))) __bf16        v16bf;
typedef __attribute__((ext_vector_type(8)))  __bf16        v8bf;
typedef __attribute__((ext_vector_type(8)))  float         v8f;
typedef __attribute__((ext_vector_type(4)))  float         v4f;
typedef __attribute__((ext_vector_type(4)))  unsigned int  u32x4;

// ---------------- problem sizes ----------------
#define N_   4
#define CIN  96
#define H_   128
#define W_   256
#define HP   130     // H + 2 (padded)
#define WP   258     // W + 2 (padded)
#define C1   128     // conv1 out channels
#define C2   144     // conv2 out channels (9 * 16)
#define K1   (9*CIN) // 864  -> 27 K-steps of 32
#define K2   (9*C1)  // 1152 -> 36 K-steps of 32

// Build a 16-element bf16 fragment from two aligned 8x16-bit chunks.
static __device__ __forceinline__ v16bf load_frag(const unsigned short* p0,
                                                  const unsigned short* p1) {
  us8 lo = *(const us8*)p0;
  us8 hi = *(const us8*)p1;
  us16 v = __builtin_shufflevector(lo, hi,
                                   0,1,2,3,4,5,6,7,8,9,10,11,12,13,14,15);
  return __builtin_bit_cast(v16bf, v);
}

// -------- pack feat (NCHW f32) -> zero-padded NHWC bf16 --------
__global__ void k_featpad(const float* __restrict__ feat,
                          unsigned short* __restrict__ fp) {
  size_t idx = (size_t)blockIdx.x * blockDim.x + threadIdx.x;
  const size_t total = (size_t)N_ * HP * WP * CIN;
  if (idx >= total) return;
  int c = (int)(idx % CIN); size_t p = idx / CIN;
  int x = (int)(p % WP); p /= WP;
  int y = (int)(p % HP); int n = (int)(p / HP);
  float v = 0.f;
  if (y >= 1 && y <= H_ && x >= 1 && x <= W_)
    v = feat[(((size_t)n * CIN + c) * H_ + (y - 1)) * W_ + (x - 1)];
  fp[idx] = __builtin_bit_cast(unsigned short, (__bf16)v);
}

// -------- pack weights (O,I,3,3 f32) -> bf16 [O][K], K = (ki*3+kj)*I + ci ----
__global__ void k_wcvt(const float* __restrict__ wsrc,
                       unsigned short* __restrict__ wdst, int O, int I) {
  int K = I * 9;
  int idx = blockIdx.x * blockDim.x + threadIdx.x;
  if (idx >= O * K) return;
  int m = idx / K, k = idx - m * K;
  int kidx = k / I, ci = k - kidx * I;
  int ki = kidx / 3, kj = kidx - ki * 3;
  float v = wsrc[(((size_t)m * I + ci) * 3 + ki) * 3 + kj];
  wdst[idx] = __builtin_bit_cast(unsigned short, (__bf16)v);
}

// -------- zero a buffer (u32 granularity) --------
__global__ void k_zero(unsigned int* __restrict__ p, size_t n32) {
  size_t i = (size_t)blockIdx.x * blockDim.x + threadIdx.x;
  if (i < n32) p[i] = 0u;
}

// -------- conv1: implicit GEMM (WMMA), bias + LeakyReLU, -> padded NHWC bf16 -
__global__ void __launch_bounds__(256) k_conv1(
    const unsigned short* __restrict__ fp,   // padded NHWC bf16 feat
    const unsigned short* __restrict__ w1b,  // [128][864] bf16
    const float* __restrict__ b1,
    unsigned short* __restrict__ hfp)        // padded NHWC bf16 hfeat
{
  int wave = (blockIdx.x * 256 + (int)threadIdx.x) >> 5;  // 0..8191
  int lane = (int)threadIdx.x & 31;
  int col = lane & 15, hi = lane >> 4;
  int w0 = (wave & 15) << 4;
  int h  = (wave >> 4) & 127;
  int n  = wave >> 11;
  int w  = w0 + col;

  // Per-lane base pointers; all per-step offsets are compile-time immediates.
  const unsigned short* bbase =
      fp + ((size_t)(n * HP + h) * WP + w) * CIN + hi * 8;
  const unsigned short* abase = w1b + (size_t)col * K1 + hi * 8;

  v8f acc[8] = {};
#pragma unroll
  for (int s = 0; s < 27; ++s) {
    const int kidx = s / 3;
    const int c0 = (s - kidx * 3) * 32;
    const int ki = kidx / 3, kj = kidx - ki * 3;
    const int boff = (ki * WP + kj) * CIN + c0;          // constant
    v16bf B = load_frag(bbase + boff, bbase + boff + 16);
    v16bf A[8];
#pragma unroll
    for (int t = 0; t < 8; ++t) {
      const int aoff = t * 16 * K1 + s * 32;             // constant
      A[t] = load_frag(abase + aoff, abase + aoff + 16);
    }
#pragma unroll
    for (int t = 0; t < 8; ++t)
      acc[t] = __builtin_amdgcn_wmma_f32_16x16x32_bf16(
          false, A[t], false, B, (short)0, acc[t], false, false);
  }

  size_t opix = ((size_t)(n * HP + h + 1)) * WP + (w + 1);
#pragma unroll
  for (int t = 0; t < 8; ++t) {
    v4f bA = *(const v4f*)(b1 + t * 16 + hi * 8);
    v4f bB = *(const v4f*)(b1 + t * 16 + hi * 8 + 4);
    v8bf o;
#pragma unroll
    for (int r = 0; r < 8; ++r) {
      float x = acc[t][r] + ((r < 4) ? bA[r] : bB[r - 4]);
      x = (x > 0.f) ? x : 0.1f * x;                 // LeakyReLU(0.1)
      o[r] = (__bf16)x;
    }
    *(u32x4*)(hfp + opix * C1 + t * 16 + hi * 8) = __builtin_bit_cast(u32x4, o);
  }
}

// -------- conv2 + softmax(9) + convex flow upsample, fused --------
__global__ void __launch_bounds__(256) k_conv2(
    const unsigned short* __restrict__ hfp,  // padded NHWC bf16 hfeat
    const unsigned short* __restrict__ w2b,  // [144][1152] bf16
    const float* __restrict__ b2,
    const float* __restrict__ flow,          // (4,2,128,256) f32
    float* __restrict__ out)                 // (4,2,512,1024) f32
{
  int wave = (blockIdx.x * 256 + (int)threadIdx.x) >> 5;
  int lane = (int)threadIdx.x & 31;
  int col = lane & 15, hi = lane >> 4;
  int w0 = (wave & 15) << 4;
  int h  = (wave >> 4) & 127;
  int n  = wave >> 11;
  int w  = w0 + col;

  const unsigned short* bbase =
      hfp + ((size_t)(n * HP + h) * WP + w) * C1 + hi * 8;
  const unsigned short* abase = w2b + (size_t)col * K2 + hi * 8;

  v8f acc[9] = {};
#pragma unroll
  for (int s = 0; s < 36; ++s) {
    const int kidx = s >> 2;
    const int c0 = (s & 3) * 32;
    const int ki = kidx / 3, kj = kidx - ki * 3;
    const int boff = (ki * WP + kj) * C1 + c0;           // constant
    v16bf B = load_frag(bbase + boff, bbase + boff + 16);
    v16bf A[9];
#pragma unroll
    for (int t = 0; t < 9; ++t) {
      const int aoff = t * 16 * K2 + s * 32;             // constant
      A[t] = load_frag(abase + aoff, abase + aoff + 16);
    }
#pragma unroll
    for (int t = 0; t < 9; ++t)
      acc[t] = __builtin_amdgcn_wmma_f32_16x16x32_bf16(
          false, A[t], false, B, (short)0, acc[t], false, false);
  }

  // mask logits = 0.25 * (conv2 + bias)
#pragma unroll
  for (int t = 0; t < 9; ++t) {
    v4f bA = *(const v4f*)(b2 + t * 16 + hi * 8);
    v4f bB = *(const v4f*)(b2 + t * 16 + hi * 8 + 4);
#pragma unroll
    for (int r = 0; r < 8; ++r)
      acc[t][r] = 0.25f * (acc[t][r] + ((r < 4) ? bA[r] : bB[r - 4]));
  }

  // 9 shifted copies of 4*flow for this pixel (zero-padded)
  float fx[9], fy[9];
#pragma unroll
  for (int t = 0; t < 9; ++t) {
    int ki = t / 3, kj = t - ki * 3;
    int hh = h + ki - 1, ww = w + kj - 1;
    int ok = (hh >= 0) & (hh < H_) & (ww >= 0) & (ww < W_);
    int hc = hh < 0 ? 0 : (hh > H_ - 1 ? H_ - 1 : hh);
    int wc = ww < 0 ? 0 : (ww > W_ - 1 ? W_ - 1 : ww);
    size_t fi = (((size_t)n * 2 + 0) * H_ + hc) * W_ + wc;
    fx[t] = ok ? 4.f * flow[fi] : 0.f;
    fy[t] = ok ? 4.f * flow[fi + (size_t)H_ * W_] : 0.f;
  }

  // softmax over the 9 taps per (a,b) slot, then convex combination
#pragma unroll
  for (int r = 0; r < 8; ++r) {
    float mx = acc[0][r];
#pragma unroll
    for (int t = 1; t < 9; ++t) mx = fmaxf(mx, acc[t][r]);
    float e[9], sum = 0.f;
#pragma unroll
    for (int t = 0; t < 9; ++t) { e[t] = __expf(acc[t][r] - mx); sum += e[t]; }
    float inv = 1.f / sum;
    float ox = 0.f, oy = 0.f;
#pragma unroll
    for (int t = 0; t < 9; ++t) { ox += e[t] * fx[t]; oy += e[t] * fy[t]; }
    ox *= inv; oy *= inv;
    int row = r + 8 * hi;          // a*4 + b
    int a = row >> 2, bb = row & 3;
    size_t oidx = (((size_t)n * 2 + 0) * (4 * H_) + (h * 4 + a)) *
                      (size_t)(4 * W_) + (w * 4 + bb);
    out[oidx] = ox;
    out[oidx + (size_t)(4 * H_) * (4 * W_)] = oy;
  }
}

// ---------------- launcher ----------------
extern "C" void kernel_launch(void* const* d_in, const int* in_sizes, int n_in,
                              void* d_out, int out_size, void* d_ws,
                              size_t ws_size, hipStream_t stream) {
  (void)in_sizes; (void)n_in; (void)out_size; (void)ws_size;
  const float* flow = (const float*)d_in[0];
  const float* feat = (const float*)d_in[1];
  const float* w1   = (const float*)d_in[2];
  const float* b1   = (const float*)d_in[3];
  const float* w2   = (const float*)d_in[4];
  const float* b2   = (const float*)d_in[5];

  char* ws = (char*)d_ws;
  const size_t o_fp  = 0;                       // 4*130*258*96 bf16 = 25,758,720 B
  const size_t o_w1  = 25758720;                // 128*864  bf16 =   221,184 B
  const size_t o_w2  = o_w1 + 221184;           // 144*1152 bf16 =   331,776 B
  const size_t o_hfp = o_w2 + 331776;           // 4*130*258*128 bf16 = 34,344,960 B
  unsigned short* fp  = (unsigned short*)(ws + o_fp);
  unsigned short* w1b = (unsigned short*)(ws + o_w1);
  unsigned short* w2b = (unsigned short*)(ws + o_w2);
  unsigned short* hfp = (unsigned short*)(ws + o_hfp);
  float* out = (float*)d_out;

  {
    size_t total = (size_t)N_ * HP * WP * CIN;
    k_featpad<<<(int)((total + 255) / 256), 256, 0, stream>>>(feat, fp);
  }
  {
    int total = C1 * K1;
    k_wcvt<<<(total + 255) / 256, 256, 0, stream>>>(w1, w1b, C1, CIN);
  }
  {
    int total = C2 * K2;
    k_wcvt<<<(total + 255) / 256, 256, 0, stream>>>(w2, w2b, C2, C1);
  }
  {
    size_t n32 = (size_t)N_ * HP * WP * C1 / 2;  // u16 count / 2
    k_zero<<<(int)((n32 + 255) / 256), 256, 0, stream>>>((unsigned int*)hfp, n32);
  }
  // 8192 wave-tiles (16 pixels each), 8 waves per 256-thread block
  k_conv1<<<1024, 256, 0, stream>>>(fp, w1b, b1, hfp);
  k_conv2<<<1024, 256, 0, stream>>>(hfp, w2b, b2, flow, out);
}
